// StarTransformerLayer_75960791597420
// MI455X (gfx1250) — matrix operand order repaired and checked by reading, not compiled, not run on hardware
//
#include <hip/hip_runtime.h>
#include <cstddef>

// ---------------- types ----------------
typedef __bf16 bf16;
typedef __attribute__((ext_vector_type(8)))  bf16  v8bf;
typedef __attribute__((ext_vector_type(16))) bf16  v16bf;
typedef __attribute__((ext_vector_type(8)))  float v8f;
typedef __attribute__((ext_vector_type(4)))  int   i32x4;

#define DD 256            // model dim
#define HH 8              // heads
#define DH 32             // head dim == wave32
#define BB 8              // batch
#define LL 8192           // seq len
#define NTOK (BB*LL)      // 65536 tokens
#define MBLK 256          // rows per workgroup (8 waves x 32 rows)
#define NBLK 64           // cols per workgroup
#define BSTRIDE 264       // LDS row stride for B slab (256 + 8 pad: kills 512B-stride bank conflicts)

static __device__ __forceinline__ float bf2f(bf16 x) { return (float)x; }

// ---- gfx1250 async global->LDS path (ASYNCcnt), with portable fallback ----
#if __has_builtin(__builtin_amdgcn_global_load_async_to_lds_b128) && __has_builtin(__builtin_amdgcn_s_wait_asynccnt)
#define HAVE_ASYNC 1
typedef __attribute__((address_space(1))) i32x4 gvec_t;   // global int4
typedef __attribute__((address_space(3))) i32x4 lvec_t;   // LDS int4
static __device__ __forceinline__ void cp16_async(const bf16* g, bf16* l) {
    __builtin_amdgcn_global_load_async_to_lds_b128((gvec_t*)g, (lvec_t*)l, 0, 0);
}
static __device__ __forceinline__ void async_wait_all() {
    __builtin_amdgcn_s_wait_asynccnt(0);
}
#else
#define HAVE_ASYNC 0
static __device__ __forceinline__ void cp16_async(const bf16* g, bf16* l) {
    *(v8bf*)l = *(const v8bf*)g;      // sync fallback: load + ds_store
}
static __device__ __forceinline__ void async_wait_all() {}
#endif

// ---------------- f32 -> bf16 convert ----------------
__global__ __launch_bounds__(256)
void cvt_f32_bf16(const float* __restrict__ in, bf16* __restrict__ out, int n) {
    int i = blockIdx.x * 256 + threadIdx.x;
    if (i < n) out[i] = (bf16)in[i];
}

// ---------------- WMMA GEMM: Y[M x 256] = X[M x 256] @ W(out,in)^T + bias ----------------
// 256 threads = 8 waves; block tile = 256(M) x 64(N); wave tile = 32(M) x 64(N).
// B slab (64 x 256 bf16, 33KB padded) staged once into LDS via async-to-LDS, shared by all 8 waves.
// Per k-step each wave runs 8 independent v_wmma_f32_16x16x32_bf16 chains (2 A-frags x 4 B-tiles).
// Requires M % MBLK == 0 (all callers use M = 65536).
__global__ __launch_bounds__(256)
void gemm_wmma_bf16(const bf16* __restrict__ A, const bf16* __restrict__ W,
                    const float* __restrict__ bias,
                    float* __restrict__ outF, bf16* __restrict__ outB, int M) {
    __shared__ __align__(16) bf16 Bs[64 * BSTRIDE];

    const int tid  = threadIdx.x;
    const int wave = tid >> 5;
    const int lane = tid & 31;
    const int sel  = lane >> 4;      // which K-half of the fragment this lane holds
    const int mi   = lane & 15;
    const int m0   = blockIdx.x * MBLK + wave * 32;   // this wave's 32 rows
    const int n0   = blockIdx.y * NBLK;

    // ---- stage B slab: W rows n0..n0+63, all 256 k -> LDS (2048 x 16B chunks, 8 per thread) ----
    for (int i = tid; i < 64 * 32; i += 256) {
        const int r = i >> 5;            // row within slab
        const int c = (i & 31) * 8;      // k offset (8 bf16 = 16B chunk)
        cp16_async(W + (size_t)(n0 + r) * DD + c, &Bs[r * BSTRIDE + c]);
    }
    async_wait_all();
    __syncthreads();

    v8f acc0[4] = {}, acc1[4] = {};      // rows 0..15 / rows 16..31 of the wave tile
    const bf16* Arow0 = A + (size_t)(m0 + mi)      * DD;
    const bf16* Arow1 = A + (size_t)(m0 + 16 + mi) * DD;

    #pragma unroll
    for (int k0 = 0; k0 < DD; k0 += 32) {
        // A fragments straight from global (per-lane two contiguous 16B chunks)
        v8bf lo0 = *(const v8bf*)(Arow0 + k0 + sel * 8);
        v8bf hi0 = *(const v8bf*)(Arow0 + k0 + 16 + sel * 8);
        v8bf lo1 = *(const v8bf*)(Arow1 + k0 + sel * 8);
        v8bf hi1 = *(const v8bf*)(Arow1 + k0 + 16 + sel * 8);
        v16bf a0 = __builtin_shufflevector(lo0, hi0, 0,1,2,3,4,5,6,7,8,9,10,11,12,13,14,15);
        v16bf a1 = __builtin_shufflevector(lo1, hi1, 0,1,2,3,4,5,6,7,8,9,10,11,12,13,14,15);

        const int kb = k0 + sel * 8;
        #pragma unroll
        for (int nt = 0; nt < 4; ++nt) {
            const bf16* bp = &Bs[(nt * 16 + mi) * BSTRIDE + kb];
            v8bf blo = *(const v8bf*)(bp);
            v8bf bhi = *(const v8bf*)(bp + 16);
            v16bf b = __builtin_shufflevector(blo, bhi, 0,1,2,3,4,5,6,7,8,9,10,11,12,13,14,15);
            acc0[nt] = __builtin_amdgcn_wmma_f32_16x16x32_bf16(false, a0, false, b, (short)0, acc0[nt], false, false);
            acc1[nt] = __builtin_amdgcn_wmma_f32_16x16x32_bf16(false, a1, false, b, (short)0, acc1[nt], false, false);
        }
    }

    // D layout: lane<16 -> N=lane, rows +0..7; lane>=16 -> N=lane-16, rows +8..15
    #pragma unroll
    for (int nt = 0; nt < 4; ++nt) {
        const int n = n0 + nt * 16 + mi;
        const float bv = bias ? bias[n] : 0.0f;
        #pragma unroll
        for (int r = 0; r < 8; ++r) {
            const int mrA = m0 + sel * 8 + r;
            const int mrB = mrA + 16;
            float vA = acc0[nt][r] + bv;
            float vB = acc1[nt][r] + bv;
            if (outF) { outF[(size_t)mrA * DD + n] = vA; outF[(size_t)mrB * DD + n] = vB; }
            if (outB) { outB[(size_t)mrA * DD + n] = (bf16)vA; outB[(size_t)mrB * DD + n] = (bf16)vB; }
        }
    }
    (void)M;
}

// ---------------- per-token 5-key star attention ----------------
// block = 256 threads = 8 waves; wave w == head w (DH==32==wave32)
__global__ __launch_bounds__(256)
void sat_attn_kernel(const bf16* __restrict__ Q,
                     const bf16* __restrict__ Kcs, const bf16* __restrict__ Vcs,
                     const bf16* __restrict__ Khs, const bf16* __restrict__ Vhs,
                     const bf16* __restrict__ Kc,  const bf16* __restrict__ Vc,
                     bf16* __restrict__ ctx) {
    const int token = blockIdx.x;
    const int b = token / LL, l = token - b * LL;
    const int d = threadIdx.x;
    const float q = bf2f(Q[(size_t)token * DD + d]);

    const int lb = (l + 1 == LL) ? 0 : l + 1;   // before = roll(cs,-1)
    const int la = (l == 0) ? (LL - 1) : 0;     // buggy forward shift
    const size_t r0 = ((size_t)(b * LL + lb)) * DD + d;
    const size_t r1 = ((size_t)token) * DD + d;
    const size_t r2 = ((size_t)(b * LL + la)) * DD + d;
    const size_t r4 = ((size_t)b) * DD + d;

    float s[5];
    s[0] = q * bf2f(Kcs[r0]);
    s[1] = q * bf2f(Kcs[r1]);
    s[2] = q * bf2f(Kcs[r2]);
    s[3] = q * bf2f(Khs[r1]);
    s[4] = q * bf2f(Kc [r4]);
    #pragma unroll
    for (int j = 0; j < 5; ++j)
        #pragma unroll
        for (int off = 16; off; off >>= 1)
            s[j] += __shfl_xor(s[j], off, 32);

    const float scale = 0.17677669529663687f;   // 1/sqrt(32)
    float mx = s[0];
    #pragma unroll
    for (int j = 1; j < 5; ++j) mx = fmaxf(mx, s[j]);
    float p[5], sum = 0.0f;
    #pragma unroll
    for (int j = 0; j < 5; ++j) { p[j] = __expf(s[j] * scale - mx * scale); sum += p[j]; }
    const float inv = 1.0f / sum;

    float c = p[0] * bf2f(Vcs[r0]) + p[1] * bf2f(Vcs[r1]) + p[2] * bf2f(Vcs[r2])
            + p[3] * bf2f(Vhs[r1]) + p[4] * bf2f(Vc [r4]);
    ctx[(size_t)token * DD + d] = (bf16)(c * inv);
}

// ---------------- fused ReLU + LayerNorm (row of 256), in-place capable ----------------
__global__ __launch_bounds__(256)
void relu_ln_kernel(const float* __restrict__ X, const float* __restrict__ w,
                    const float* __restrict__ bia, float* __restrict__ outF,
                    bf16* __restrict__ outB) {
    __shared__ float red[256];
    const int row = blockIdx.x, t = threadIdx.x;
    float x = X[(size_t)row * DD + t];
    x = x > 0.0f ? x : 0.0f;
    red[t] = x; __syncthreads();
    for (int s = 128; s; s >>= 1) { if (t < s) red[t] += red[t + s]; __syncthreads(); }
    const float mean = red[0] * (1.0f / DD); __syncthreads();
    const float dx = x - mean;
    red[t] = dx * dx; __syncthreads();
    for (int s = 128; s; s >>= 1) { if (t < s) red[t] += red[t + s]; __syncthreads(); }
    const float var = red[0] * (1.0f / DD);
    const float y = w[t] * (dx * rsqrtf(var + 1e-12f)) + bia[t];
    if (outF) outF[(size_t)row * DD + t] = y;
    if (outB) outB[(size_t)row * DD + t] = (bf16)y;
}

// ---------------- relay attention: per (b,h), softmax over L+1 keys ----------------
__global__ __launch_bounds__(256)
void rel_attn_kernel(const float* __restrict__ qrel,
                     const bf16* __restrict__ Kr, const bf16* __restrict__ Vr,
                     const bf16* __restrict__ Kc, const bf16* __restrict__ Vc,
                     float* __restrict__ ctx) {
    __shared__ float sc[LL + 1];
    __shared__ float red[256];
    __shared__ float qsh[DH];
    __shared__ float part[256];
    const int b = blockIdx.x >> 3, h = blockIdx.x & 7;
    const int t = threadIdx.x;
    const int Tk = LL + 1;
    if (t < DH) qsh[t] = qrel[(size_t)b * DD + h * DH + t];
    __syncthreads();

    const float scale = 0.17677669529663687f;
    float lmax = -3.4e38f;
    for (int k = t; k < Tk; k += 256) {
        const bf16* kp = (k == 0) ? (Kc + (size_t)b * DD + h * DH)
                                  : (Kr + ((size_t)b * LL + (k - 1)) * DD + h * DH);
        float s = 0.0f;
        #pragma unroll
        for (int j = 0; j < DH; ++j) s += qsh[j] * bf2f(kp[j]);
        s *= scale;
        sc[k] = s;
        lmax = fmaxf(lmax, s);
    }
    red[t] = lmax; __syncthreads();
    for (int s = 128; s; s >>= 1) { if (t < s) red[t] = fmaxf(red[t], red[t + s]); __syncthreads(); }
    const float mx = red[0]; __syncthreads();
    float lsum = 0.0f;
    for (int k = t; k < Tk; k += 256) { float p = __expf(sc[k] - mx); sc[k] = p; lsum += p; }
    red[t] = lsum; __syncthreads();
    for (int s = 128; s; s >>= 1) { if (t < s) red[t] += red[t + s]; __syncthreads(); }
    const float inv = 1.0f / red[0];
    __syncthreads();

    const int dd = t & 31, g = t >> 5;
    float acc = 0.0f;
    for (int k = g; k < Tk; k += 8) {
        const bf16* vp = (k == 0) ? (Vc + (size_t)b * DD + h * DH)
                                  : (Vr + ((size_t)b * LL + (k - 1)) * DD + h * DH);
        acc += sc[k] * bf2f(vp[dd]);
    }
    part[t] = acc; __syncthreads();
    for (int s = 4; s; s >>= 1) { if (g < s) part[t] += part[(g + s) * 32 + dd]; __syncthreads(); }
    if (g == 0) ctx[(size_t)b * DD + h * DH + dd] = part[dd] * inv;
}

// ---------------- tiny projections for 8-row operands ----------------
__global__ __launch_bounds__(256)
void small_proj_kernel(const float* __restrict__ X, const float* __restrict__ W,
                       const float* __restrict__ bias,
                       float* __restrict__ outF, bf16* __restrict__ outB) {
    const int row = blockIdx.x, n = threadIdx.x;
    const float* x = X + (size_t)row * DD;
    const float* w = W + (size_t)n * DD;
    float s = bias[n];
    for (int k = 0; k < DD; ++k) s += x[k] * w[k];
    if (outF) outF[(size_t)row * DD + n] = s;
    if (outB) outB[(size_t)row * DD + n] = (bf16)s;
}

// ---------------- relay out-proj + ReLU + LN -> center ----------------
__global__ __launch_bounds__(256)
void rel_out_kernel(const float* __restrict__ ctx, const float* __restrict__ Wo,
                    const float* __restrict__ bo, const float* __restrict__ lnw,
                    const float* __restrict__ lnb, float* __restrict__ centerOut) {
    __shared__ float red[256];
    const int b = blockIdx.x, t = threadIdx.x;
    const float* x = ctx + (size_t)b * DD;
    const float* w = Wo + (size_t)t * DD;
    float s = bo[t];
    for (int k = 0; k < DD; ++k) s += x[k] * w[k];
    s = s > 0.0f ? s : 0.0f;
    red[t] = s; __syncthreads();
    for (int q = 128; q; q >>= 1) { if (t < q) red[t] += red[t + q]; __syncthreads(); }
    const float mean = red[0] * (1.0f / DD); __syncthreads();
    const float dx = s - mean;
    red[t] = dx * dx; __syncthreads();
    for (int q = 128; q; q >>= 1) { if (t < q) red[t] += red[t + q]; __syncthreads(); }
    const float var = red[0] * (1.0f / DD);
    centerOut[(size_t)b * DD + t] = lnw[t] * (dx * rsqrtf(var + 1e-12f)) + lnb[t];
}

// ---------------- host orchestration ----------------
extern "C" void kernel_launch(void* const* d_in, const int* in_sizes, int n_in,
                              void* d_out, int out_size, void* d_ws, size_t ws_size,
                              hipStream_t stream) {
    const float* hs      = (const float*)d_in[0];
    const float* center0 = (const float*)d_in[1];
    const float* sat_W[4] = {(const float*)d_in[2], (const float*)d_in[3], (const float*)d_in[4], (const float*)d_in[5]};
    const float* sat_b[4] = {(const float*)d_in[6], (const float*)d_in[7], (const float*)d_in[8], (const float*)d_in[9]};
    const float* rel_W[4] = {(const float*)d_in[10], (const float*)d_in[11], (const float*)d_in[12], (const float*)d_in[13]};
    const float* rel_b[4] = {(const float*)d_in[14], (const float*)d_in[15], (const float*)d_in[16], (const float*)d_in[17]};
    const float* ln_sat_w = (const float*)d_in[18];
    const float* ln_sat_b = (const float*)d_in[19];
    const float* ln_rel_w = (const float*)d_in[20];
    const float* ln_rel_b = (const float*)d_in[21];

    float* cs_out     = (float*)d_out;                  // (B,L,D) f32; also used as GEMM f32 scratch
    float* center_out = (float*)d_out + (size_t)NTOK * DD;

    // ---- carve workspace ----
    size_t off = 0;
    auto alloc = [&](size_t bytes) -> void* {
        void* p = (char*)d_ws + off;
        off = (off + bytes + 255) & ~(size_t)255;
        return p;
    };
    const size_t TOKB = (size_t)NTOK * DD * sizeof(bf16);   // 32 MiB
    bf16* hs_b   = (bf16*)alloc(TOKB);
    bf16* Khs    = (bf16*)alloc(TOKB);
    bf16* Vhs    = (bf16*)alloc(TOKB);
    bf16* Qcs    = (bf16*)alloc(TOKB);
    bf16* Kcs    = (bf16*)alloc(TOKB);
    bf16* Vcs    = (bf16*)alloc(TOKB);
    bf16* ctxb   = (bf16*)alloc(TOKB);
    bf16* csb    = (bf16*)alloc(TOKB);
    bf16* wsatq  = (bf16*)alloc(DD * DD * sizeof(bf16));
    bf16* wsatk  = (bf16*)alloc(DD * DD * sizeof(bf16));
    bf16* wsatv  = (bf16*)alloc(DD * DD * sizeof(bf16));
    bf16* wsato  = (bf16*)alloc(DD * DD * sizeof(bf16));
    bf16* wrelk  = (bf16*)alloc(DD * DD * sizeof(bf16));
    bf16* wrelv  = (bf16*)alloc(DD * DD * sizeof(bf16));
    bf16* Kc_sat = (bf16*)alloc(BB * DD * sizeof(bf16));
    bf16* Vc_sat = (bf16*)alloc(BB * DD * sizeof(bf16));
    bf16* Kc_rel = (bf16*)alloc(BB * DD * sizeof(bf16));
    bf16* Vc_rel = (bf16*)alloc(BB * DD * sizeof(bf16));
    float* qrel    = (float*)alloc(BB * DD * sizeof(float));
    float* ctxrel  = (float*)alloc(BB * DD * sizeof(float));
    float* centerA = (float*)alloc(BB * DD * sizeof(float));
    (void)ws_size; (void)n_in; (void)in_sizes; (void)out_size;

    auto cvt = [&](const float* src, bf16* dst, int n) {
        cvt_f32_bf16<<<(n + 255) / 256, 256, 0, stream>>>(src, dst, n);
    };
    auto gemm = [&](const bf16* A, const bf16* W, const float* bias,
                    float* oF, bf16* oB, int M) {
        dim3 grid(M / MBLK, DD / NBLK);                 // M % 256 == 0 for all callers
        gemm_wmma_bf16<<<grid, 256, 0, stream>>>(A, W, bias, oF, oB, M);
    };

    // ---- one-time prep ----
    cvt(hs, hs_b, NTOK * DD);
    cvt(sat_W[0], wsatq, DD * DD);
    cvt(sat_W[1], wsatk, DD * DD);
    cvt(sat_W[2], wsatv, DD * DD);
    cvt(sat_W[3], wsato, DD * DD);
    cvt(rel_W[1], wrelk, DD * DD);
    cvt(rel_W[2], wrelv, DD * DD);
    gemm(hs_b, wsatk, sat_b[1], nullptr, Khs, NTOK);   // K(hidden) with sat weights, reused both iters
    gemm(hs_b, wsatv, sat_b[2], nullptr, Vhs, NTOK);

    const float* center_cur = center0;
    for (int it = 0; it < 2; ++it) {
        const bf16* cs_cur = (it == 0) ? hs_b : csb;
        const bf16* Kuse, *Vuse;
        if (it == 0) { Kuse = Khs; Vuse = Vhs; }       // cs == hidden on iter 0
        else {
            gemm(cs_cur, wsatk, sat_b[1], nullptr, Kcs, NTOK);
            gemm(cs_cur, wsatv, sat_b[2], nullptr, Vcs, NTOK);
            Kuse = Kcs; Vuse = Vcs;
        }
        gemm(cs_cur, wsatq, sat_b[0], nullptr, Qcs, NTOK);
        small_proj_kernel<<<BB, 256, 0, stream>>>(center_cur, sat_W[1], sat_b[1], nullptr, Kc_sat);
        small_proj_kernel<<<BB, 256, 0, stream>>>(center_cur, sat_W[2], sat_b[2], nullptr, Vc_sat);

        sat_attn_kernel<<<NTOK, 256, 0, stream>>>(Qcs, Kuse, Vuse, Khs, Vhs, Kc_sat, Vc_sat, ctxb);
        gemm(ctxb, wsato, sat_b[3], cs_out, nullptr, NTOK);         // att -> d_out (f32)
        relu_ln_kernel<<<NTOK, 256, 0, stream>>>(cs_out, ln_sat_w, ln_sat_b, cs_out, csb);

        // relay attention
        gemm(csb, wrelk, rel_b[1], nullptr, Kcs, NTOK);             // reuse Kcs/Vcs buffers
        gemm(csb, wrelv, rel_b[2], nullptr, Vcs, NTOK);
        small_proj_kernel<<<BB, 256, 0, stream>>>(center_cur, rel_W[1], rel_b[1], nullptr, Kc_rel);
        small_proj_kernel<<<BB, 256, 0, stream>>>(center_cur, rel_W[2], rel_b[2], nullptr, Vc_rel);
        small_proj_kernel<<<BB, 256, 0, stream>>>(center_cur, rel_W[0], rel_b[0], qrel, nullptr);
        rel_attn_kernel<<<BB * HH, 256, 0, stream>>>(qrel, Kcs, Vcs, Kc_rel, Vc_rel, ctxrel);
        float* cdst = (it == 1) ? center_out : centerA;
        rel_out_kernel<<<BB, 256, 0, stream>>>(ctxrel, rel_W[3], rel_b[3], ln_rel_w, ln_rel_b, cdst);
        center_cur = cdst;
    }
}
